// PerceiverAttention_mask_25237227832050
// MI455X (gfx1250) — matrix operand order repaired and checked by reading, not compiled, not run on hardware
//
#include <hip/hip_runtime.h>

// ---------------------------------------------------------------------------
// PerceiverAttention (masked) for gfx1250: all GEMMs via v_wmma_f32_16x16x32_f16
// b=8 t=4 n=2048 nl=64 dim=1024 heads=8 dhead=64 inner=512, NJ = n+nl = 2112
// ---------------------------------------------------------------------------

typedef _Float16 f16;
typedef _Float16 h4_t  __attribute__((ext_vector_type(4)));
typedef _Float16 h8_t  __attribute__((ext_vector_type(8)));
typedef _Float16 v16h  __attribute__((ext_vector_type(16)));
typedef float    v8f   __attribute__((ext_vector_type(8)));

#define CAT16(lo, hi) __builtin_shufflevector(lo, hi, 0,1,2,3,4,5,6,7,8,9,10,11,12,13,14,15)

// ---------------------- weight convert + transpose -------------------------
// W[K][N] (f32, row-major) -> WT[N][K] (f16, row-major) so B^T rows are
// K-contiguous for WMMA B-fragment b128 loads.
__global__ __launch_bounds__(256) void cvt_transpose_kernel(
    const float* __restrict__ W, f16* __restrict__ WT, int K, int N)
{
    int idx = blockIdx.x * 256 + threadIdx.x;
    if (idx < K * N) {
        int k = idx / N, n = idx % N;
        WT[(size_t)n * K + k] = (f16)W[idx];
    }
}

// ------------------------------ layernorm ----------------------------------
__device__ __forceinline__ void ln_row_body(
    const float* __restrict__ src, const float* __restrict__ w,
    const float* __restrict__ b, f16* __restrict__ dst1, f16* __restrict__ dst2)
{
    const int tid = threadIdx.x;           // 256 threads, 1024 elems, 4/thread
    __shared__ float red[256];
    float4 v = *(const float4*)(src + tid * 4);
    red[tid] = v.x + v.y + v.z + v.w;
    __syncthreads();
    for (int off = 128; off > 0; off >>= 1) {
        if (tid < off) red[tid] += red[tid + off];
        __syncthreads();
    }
    float mean = red[0] * (1.0f / 1024.0f);
    __syncthreads();
    float d0 = v.x - mean, d1 = v.y - mean, d2 = v.z - mean, d3 = v.w - mean;
    red[tid] = d0 * d0 + d1 * d1 + d2 * d2 + d3 * d3;
    __syncthreads();
    for (int off = 128; off > 0; off >>= 1) {
        if (tid < off) red[tid] += red[tid + off];
        __syncthreads();
    }
    float rs = rsqrtf(red[0] * (1.0f / 1024.0f) + 1e-5f);
    float4 wv = *(const float4*)(w + tid * 4);
    float4 bv = *(const float4*)(b + tid * 4);
    h4_t o;
    o[0] = (f16)(d0 * rs * wv.x + bv.x);
    o[1] = (f16)(d1 * rs * wv.y + bv.y);
    o[2] = (f16)(d2 * rs * wv.z + bv.z);
    o[3] = (f16)(d3 * rs * wv.w + bv.w);
    *(h4_t*)(dst1 + tid * 4) = o;
    if (dst2) *(h4_t*)(dst2 + tid * 4) = o;
}

// x (b,t,2048,1024) -> kvin rows 0..2047 of each (b,t) block of 2112 rows
__global__ __launch_bounds__(256) void ln_x_kernel(
    const float* __restrict__ x, const float* __restrict__ w,
    const float* __restrict__ b, f16* __restrict__ kvin)
{
    int r = blockIdx.x;                     // bt*2048 + j
    int bt = r >> 11, j = r & 2047;
    ln_row_body(x + (size_t)r * 1024, w, b,
                kvin + ((size_t)bt * 2112 + j) * 1024, nullptr);
}

// latents (b,t,64,1024) -> kvin rows 2048..2111 AND contiguous qin
__global__ __launch_bounds__(256) void ln_lat_kernel(
    const float* __restrict__ lat, const float* __restrict__ w,
    const float* __restrict__ b, f16* __restrict__ kvin, f16* __restrict__ qin)
{
    int r = blockIdx.x;                     // bt*64 + i
    int bt = r >> 6, i = r & 63;
    ln_row_body(lat + (size_t)r * 1024, w, b,
                kvin + ((size_t)bt * 2112 + 2048 + i) * 1024,
                qin + (size_t)r * 1024);
}

// ------------------------- generic batched WMMA GEMM -----------------------
// C[z] = A[z] (MxK, f16 row-major) * Bt[z]^T (Bt is NxK, f16 row-major)
// Wave tile 16x64 (4 accumulators), workgroup = 8 waves -> 64(M) x 128(N).
// Double-buffered K loop: loads for k+32 are issued before the WMMAs for k.
// MODE 0: C f16 row-major (scaled). MODE 1: C^T f16 (N x M). MODE 2: C f32.
template <int MODE>
__global__ __launch_bounds__(256) void gemm_wmma(
    const f16* __restrict__ A,  long long lda, long long sAi, long long sAo,
    const f16* __restrict__ Bt, long long ldb, long long sBi, long long sBo,
    void* __restrict__ Cv,      long long ldc, long long sCi, long long sCo,
    int M, int N, int K, int innerBatch, float cScale)
{
    const int z  = blockIdx.z;
    const int zo = z / innerBatch, zi = z % innerBatch;
    A  += (size_t)zo * sAo + (size_t)zi * sAi;
    Bt += (size_t)zo * sBo + (size_t)zi * sBi;

    const int wave = threadIdx.x >> 5;
    const int lane = threadIdx.x & 31;
    const int l    = lane & 15;
    const bool hi  = lane >= 16;
    const int wm = wave & 3, wn = wave >> 2;
    const int m0 = blockIdx.x * 64 + wm * 16;
    const int n0 = blockIdx.y * 128 + wn * 64;

    // A fragment rows: lane<16 gets K 0..15, lane>=16 gets K 16..31
    int ar = m0 + l; if (ar > M - 1) ar = M - 1;
    const f16* ap = A + (size_t)ar * lda + (hi ? 16 : 0);

    // B fragment: lane<16 holds K 0..15 of col n, lane>=16 holds K 16..31
    // (scalar pointer variables, NOT an array -> keeps global addrspace inference)
    int bn0 = n0 + 0  + l; if (bn0 > N - 1) bn0 = N - 1;
    int bn1 = n0 + 16 + l; if (bn1 > N - 1) bn1 = N - 1;
    int bn2 = n0 + 32 + l; if (bn2 > N - 1) bn2 = N - 1;
    int bn3 = n0 + 48 + l; if (bn3 > N - 1) bn3 = N - 1;
    const int boff = hi ? 16 : 0;
    const f16* bp0 = Bt + (size_t)bn0 * ldb + boff;
    const f16* bp1 = Bt + (size_t)bn1 * ldb + boff;
    const f16* bp2 = Bt + (size_t)bn2 * ldb + boff;
    const f16* bp3 = Bt + (size_t)bn3 * ldb + boff;

    v8f acc0, acc1, acc2, acc3;
#pragma unroll
    for (int r = 0; r < 8; ++r) { acc0[r] = 0.f; acc1[r] = 0.f; acc2[r] = 0.f; acc3[r] = 0.f; }

    // ---- preload k-step 0 ----
    h8_t a0  = *(const h8_t*)(ap);       h8_t a1  = *(const h8_t*)(ap + 8);
    h8_t b0l = *(const h8_t*)(bp0);      h8_t b0h = *(const h8_t*)(bp0 + 8);
    h8_t b1l = *(const h8_t*)(bp1);      h8_t b1h = *(const h8_t*)(bp1 + 8);
    h8_t b2l = *(const h8_t*)(bp2);      h8_t b2h = *(const h8_t*)(bp2 + 8);
    h8_t b3l = *(const h8_t*)(bp3);      h8_t b3h = *(const h8_t*)(bp3 + 8);
    ap += 32; bp0 += 32; bp1 += 32; bp2 += 32; bp3 += 32;

    const int steps = K >> 5;
    for (int s = 0; s < steps - 1; ++s) {
        // issue next k-step loads before consuming current fragments
        h8_t na0  = *(const h8_t*)(ap);   h8_t na1  = *(const h8_t*)(ap + 8);
        h8_t nb0l = *(const h8_t*)(bp0);  h8_t nb0h = *(const h8_t*)(bp0 + 8);
        h8_t nb1l = *(const h8_t*)(bp1);  h8_t nb1h = *(const h8_t*)(bp1 + 8);
        h8_t nb2l = *(const h8_t*)(bp2);  h8_t nb2h = *(const h8_t*)(bp2 + 8);
        h8_t nb3l = *(const h8_t*)(bp3);  h8_t nb3h = *(const h8_t*)(bp3 + 8);
        ap += 32; bp0 += 32; bp1 += 32; bp2 += 32; bp3 += 32;

        v16h aF = CAT16(a0, a1);
        acc0 = __builtin_amdgcn_wmma_f32_16x16x32_f16(false, aF, false, CAT16(b0l, b0h), (short)0, acc0, false, false);
        acc1 = __builtin_amdgcn_wmma_f32_16x16x32_f16(false, aF, false, CAT16(b1l, b1h), (short)0, acc1, false, false);
        acc2 = __builtin_amdgcn_wmma_f32_16x16x32_f16(false, aF, false, CAT16(b2l, b2h), (short)0, acc2, false, false);
        acc3 = __builtin_amdgcn_wmma_f32_16x16x32_f16(false, aF, false, CAT16(b3l, b3h), (short)0, acc3, false, false);

        a0 = na0; a1 = na1;
        b0l = nb0l; b0h = nb0h; b1l = nb1l; b1h = nb1h;
        b2l = nb2l; b2h = nb2h; b3l = nb3l; b3h = nb3h;
    }
    {   // ---- final k-step ----
        v16h aF = CAT16(a0, a1);
        acc0 = __builtin_amdgcn_wmma_f32_16x16x32_f16(false, aF, false, CAT16(b0l, b0h), (short)0, acc0, false, false);
        acc1 = __builtin_amdgcn_wmma_f32_16x16x32_f16(false, aF, false, CAT16(b1l, b1h), (short)0, acc1, false, false);
        acc2 = __builtin_amdgcn_wmma_f32_16x16x32_f16(false, aF, false, CAT16(b2l, b2h), (short)0, acc2, false, false);
        acc3 = __builtin_amdgcn_wmma_f32_16x16x32_f16(false, aF, false, CAT16(b3l, b3h), (short)0, acc3, false, false);
    }

    v8f accs[4] = {acc0, acc1, acc2, acc3};
    const int mBase = m0 + (hi ? 8 : 0);   // C layout: lane holds col n, rows mBase..+7
    if (MODE == 0) {
        f16* C = (f16*)Cv + (size_t)zo * sCo + (size_t)zi * sCi;
#pragma unroll
        for (int j = 0; j < 4; ++j) {
            int col = n0 + j * 16 + l;
            if (col < N) {
#pragma unroll
                for (int r = 0; r < 8; ++r) {
                    int row = mBase + r;
                    if (row < M) C[(size_t)row * ldc + col] = (f16)(accs[j][r] * cScale);
                }
            }
        }
    } else if (MODE == 1) {                // transposed f16 store: CT[N][M]
        f16* C = (f16*)Cv + (size_t)zo * sCo + (size_t)zi * sCi;
#pragma unroll
        for (int j = 0; j < 4; ++j) {
            int rowT = n0 + j * 16 + l;
            if (rowT < N) {
                h8_t hv;
#pragma unroll
                for (int r = 0; r < 8; ++r) hv[r] = (f16)(accs[j][r] * cScale);
                if (mBase + 7 < M) {
                    *(h8_t*)(C + (size_t)rowT * ldc + mBase) = hv;   // 16B store
                } else {
                    for (int r = 0; r < 8; ++r)
                        if (mBase + r < M) C[(size_t)rowT * ldc + mBase + r] = hv[r];
                }
            }
        }
    } else {                               // MODE 2: f32 row-major
        float* C = (float*)Cv + (size_t)zo * sCo + (size_t)zi * sCi;
#pragma unroll
        for (int j = 0; j < 4; ++j) {
            int col = n0 + j * 16 + l;
            if (col < N) {
#pragma unroll
                for (int r = 0; r < 8; ++r) {
                    int row = mBase + r;
                    if (row < M) C[(size_t)row * ldc + col] = accs[j][r] * cScale;
                }
            }
        }
    }
}

// --------------------------- masked softmax --------------------------------
// In-place over sim f16, one block per attention row (bth*64+i), length 2112.
// mask[b][j]==1 => masked (-inf) for j<2048; latent cols 2048..2111 never masked.
__global__ __launch_bounds__(256) void softmax_mask_kernel(
    f16* __restrict__ sim, const int* __restrict__ mask)
{
    const int row = blockIdx.x;            // 0..16383
    const int bth = row >> 6;              // bt*8+h ; bt=b*4+t -> b = bth/32
    const int b   = bth >> 5;
    f16* p = sim + (size_t)row * 2112;
    const int* mrow = mask + (size_t)b * 2048;
    const int tid = threadIdx.x;
    __shared__ float red[256];

    float mx = -3.0e38f;
    for (int j = tid; j < 2112; j += 256) {
        bool masked = (j < 2048) && (mrow[j] == 1);
        float v = masked ? -3.0e38f : (float)p[j];
        mx = fmaxf(mx, v);
    }
    red[tid] = mx; __syncthreads();
    for (int off = 128; off > 0; off >>= 1) {
        if (tid < off) red[tid] = fmaxf(red[tid], red[tid + off]);
        __syncthreads();
    }
    mx = red[0]; __syncthreads();

    float sum = 0.0f;
    for (int j = tid; j < 2112; j += 256) {
        bool masked = (j < 2048) && (mrow[j] == 1);
        float e = masked ? 0.0f : __expf((float)p[j] - mx);
        sum += e;
        p[j] = (f16)e;
    }
    red[tid] = sum; __syncthreads();
    for (int off = 128; off > 0; off >>= 1) {
        if (tid < off) red[tid] += red[tid + off];
        __syncthreads();
    }
    float inv = 1.0f / red[0]; __syncthreads();
    for (int j = tid; j < 2112; j += 256)
        p[j] = (f16)((float)p[j] * inv);
}

// ------------------------------- launcher ----------------------------------
extern "C" void kernel_launch(void* const* d_in, const int* in_sizes, int n_in,
                              void* d_out, int out_size, void* d_ws, size_t ws_size,
                              hipStream_t stream)
{
    (void)in_sizes; (void)n_in; (void)out_size; (void)ws_size;
    const float* x      = (const float*)d_in[0];
    const float* lat    = (const float*)d_in[1];
    const int*   mask   = (const int*)d_in[2];
    const float* ln_m_w = (const float*)d_in[3];
    const float* ln_m_b = (const float*)d_in[4];
    const float* ln_l_w = (const float*)d_in[5];
    const float* ln_l_b = (const float*)d_in[6];
    const float* Wq     = (const float*)d_in[7];
    const float* Wkv    = (const float*)d_in[8];
    const float* Wout   = (const float*)d_in[9];
    float* out = (float*)d_out;

    // workspace carve-up (halves); total ~289 MB
    char* ws = (char*)d_ws;
    size_t off = 0;
    auto carve = [&](size_t halves) -> f16* {
        f16* p = (f16*)(ws + off); off += halves * sizeof(f16); return p;
    };
    f16* kvin  = carve(32ULL * 2112 * 1024);   // LN(x)+LN(lat) f16, (bt,2112,1024)
    f16* qin   = carve(2048ULL * 1024);        // LN(lat) contiguous
    f16* WqT   = carve(512ULL * 1024);         // Wq^T   (N=512 x K=1024)
    f16* WkvT  = carve(1024ULL * 1024);        // Wkv^T  (N=1024 x K=1024)
    f16* WoutT = carve(1024ULL * 512);         // Wout^T (N=1024 x K=512)
    f16* q_h   = carve(2048ULL * 512);         // q * 0.125, (bt*64, 512)
    f16* k_h   = carve(32ULL * 2112 * 512);    // k rows d-contiguous
    f16* vT    = carve(32ULL * 512 * 2112);    // v^T: (bt, hd, j) j-contiguous
    f16* o_h   = carve(2048ULL * 512);         // attention output, heads merged
    f16* attn  = kvin;                         // reuse kvin region for sim/attn

    cvt_transpose_kernel<<<2048, 256, 0, stream>>>(Wq,   WqT,   1024, 512);
    cvt_transpose_kernel<<<4096, 256, 0, stream>>>(Wkv,  WkvT,  1024, 1024);
    cvt_transpose_kernel<<<2048, 256, 0, stream>>>(Wout, WoutT, 512,  1024);

    ln_x_kernel  <<<65536, 256, 0, stream>>>(x,   ln_m_w, ln_m_b, kvin);
    ln_lat_kernel<<<2048,  256, 0, stream>>>(lat, ln_l_w, ln_l_b, kvin, qin);

    // k = kvin @ Wkv[:, :512]   (batched over bt=32, M=2112, N=512, K=1024)
    gemm_wmma<0><<<dim3(33, 4, 32), 256, 0, stream>>>(
        kvin, 1024, 0, 2112LL * 1024,  WkvT, 1024, 0, 0,
        k_h, 512, 0, 2112LL * 512,  2112, 512, 1024, 1, 1.0f);
    // v^T (transposed store) from kvin @ Wkv[:, 512:]
    gemm_wmma<1><<<dim3(33, 4, 32), 256, 0, stream>>>(
        kvin, 1024, 0, 2112LL * 1024,  WkvT + 512 * 1024, 1024, 0, 0,
        vT, 2112, 0, 512LL * 2112,  2112, 512, 1024, 1, 1.0f);
    // q = lat_ln @ Wq, scaled by dhead^-0.5 = 0.125
    gemm_wmma<0><<<dim3(32, 4, 1), 256, 0, stream>>>(
        qin, 1024, 0, 0,  WqT, 1024, 0, 0,
        q_h, 512, 0, 0,  2048, 512, 1024, 1, 0.125f);
    // sim[bth] = q_bth (64x64) @ k_bth^T  -> (64 x 2112), z = bt*8 + h
    gemm_wmma<0><<<dim3(1, 17, 256), 256, 0, stream>>>(
        q_h, 512, 64, 64LL * 512,  k_h, 512, 64, 2112LL * 512,
        attn, 2112, 64LL * 2112, 8LL * 64 * 2112,  64, 2112, 64, 8, 1.0f);

    softmax_mask_kernel<<<16384, 256, 0, stream>>>(attn, mask);

    // out[bth] = attn (64x2112) @ v -> (64 x 64), scattered into (bt*64+i, h*64+d)
    gemm_wmma<0><<<dim3(1, 1, 256), 256, 0, stream>>>(
        attn, 2112, 64LL * 2112, 8LL * 64 * 2112,  vT, 2112, 64LL * 2112, 512LL * 2112,
        o_h, 512, 64, 64LL * 512,  64, 64, 2112, 8, 1.0f);
    // final: o (2048x512) @ Wout -> d_out f32 (2048x1024)
    gemm_wmma<2><<<dim3(32, 8, 1), 256, 0, stream>>>(
        o_h, 512, 0, 0,  WoutT, 512, 0, 0,
        out, 1024, 0, 0,  2048, 1024, 512, 1, 1.0f);
}
